// SIEGLinkPredictor_893353198384
// MI455X (gfx1250) — matrix operand order repaired and checked by reading, not compiled, not run on hardware
//
#include <hip/hip_runtime.h>

// ---------------------------------------------------------------------------
// SIEG link predictor for MI455X (gfx1250, wave32, WMMA).
// Dense GEMMs run through v_wmma_f32_16x16x32_f16 (fp32 -> f16 on the LDS
// staging path, f32 accumulation). Tiles are staged into LDS in WMMA
// *fragment order* (each lane reads its v16h operand as 2x ds_load_b128),
// double-buffered so next-tile global loads overlap WMMA execution with one
// barrier per K-step. Staging is phase-split (batch all global loads, then
// convert+store) so loads drain with pipelined s_wait_loadcnt instead of one
// full wait per element. Full-tile blocks take guard-free staging/epilogue
// paths via one uniform branch. The 4 WMMAs per step form an operand reuse
// chain (matrix_a_reuse / matrix_b_reuse). Graph aggregation uses
// global_atomic_add_f32. The 200k x 768 concat is never materialized:
// zc @ m_w1 = three accumulating (gathered) GEMMs vs the m_w1 panels.
//
// Workspace requirement: (2N + 3*N*512 + 3*P*256) floats ~= 676 MB.
// ---------------------------------------------------------------------------

typedef __attribute__((ext_vector_type(16))) _Float16 v16h;
typedef __attribute__((ext_vector_type(2)))  _Float16 v2h;
typedef __attribute__((ext_vector_type(8)))  float    v8f;

#define GEMM_TM 128
#define GEMM_TN 64
#define GEMM_TK 32
#define A_FRAG_HALVES (8 * 32 * 16)   // 4096 halves = 8 KB
#define B_FRAG_HALVES (4 * 32 * 16)   // 2048 halves = 4 KB
#define A_ITEMS 8                     // 2048 k-pairs / 256 threads
#define B_ITEMS 4                     // 1024 n-pairs / 256 threads

// Stage one 128x32 A tile and one 32x64 B tile (f32 -> f16) into LDS in
// fragment order.  A inverse map: hf=(k>>3)&1, e=(k&7)|((k&16)>>1),
// lane=m+16*hf.  B inverse map: hf=(kr>>3)&1, e=(kr&7)|((kr&16)>>1), lane=n+16*hf.
// Phase-split: all global loads first (batched, pipelined waits), then
// convert + LDS stores (indices recomputed to keep register pressure low).
template <bool GATHER, bool FULL>
__device__ __forceinline__
void sieg_stage(const float* __restrict__ A, const float* __restrict__ B,
                _Float16* __restrict__ AsF, _Float16* __restrict__ BsF,
                int tileM, int tileN, int k0, int M, int lda, int ldb,
                const int* __restrict__ rowIdx, int idxStride) {
  float fa0[A_ITEMS], fa1[A_ITEMS];
  float fb0[B_ITEMS], fb1[B_ITEMS];

  // ---- Load phase: issue every global load before any LDS store ----------
#pragma unroll
  for (int it = 0; it < A_ITEMS; ++it) {
    int i  = threadIdx.x + (it << 8);
    int r  = i >> 4;                    // row in tile (0..127)
    int c  = (i & 15) << 1;             // k in tile (even pair)
    int gr = tileM + r;
    float f0 = 0.0f, f1 = 0.0f;
    if (FULL || gr < M) {
      int row = GATHER ? rowIdx[(size_t)gr * idxStride] : gr;
      const float* ap = &A[(size_t)row * lda + (k0 + c)];
      f0 = ap[0]; f1 = ap[1];
    }
    fa0[it] = f0; fa1[it] = f1;
  }
#pragma unroll
  for (int it = 0; it < B_ITEMS; ++it) {
    int i  = threadIdx.x + (it << 8);
    int kr = i >> 5;                    // k in tile (0..31)
    int c  = (i & 31) << 1;             // n pair in tile (even)
    const float* bp = &B[(size_t)(k0 + kr) * ldb + (tileN + c)];
    fb0[it] = bp[0]; fb1[it] = bp[1];
  }

  // ---- Convert + store phase ---------------------------------------------
#pragma unroll
  for (int it = 0; it < A_ITEMS; ++it) {
    int i   = threadIdx.x + (it << 8);
    int r   = i >> 4;
    int c   = (i & 15) << 1;
    int ahf = (c >> 3) & 1;
    int e   = (c & 7) | ((c & 16) >> 1);
    int idx = (((r >> 4) << 5) + (r & 15) + (ahf << 4)) * 16 + e;
    v2h pr; pr[0] = (_Float16)fa0[it]; pr[1] = (_Float16)fa1[it];
    *reinterpret_cast<v2h*>(&AsF[idx]) = pr;
  }
#pragma unroll
  for (int it = 0; it < B_ITEMS; ++it) {
    int i  = threadIdx.x + (it << 8);
    int kr = i >> 5;
    int c  = (i & 31) << 1;
    int base = (((kr >> 3) & 1) << 4) * 16 + ((kr & 7) | ((kr & 16) >> 1));      // hf, e part
    int i0 = (((c >> 4) << 5) + (c & 15)) * 16 + base;
    int c1 = c + 1;
    int i1 = (((c1 >> 4) << 5) + (c1 & 15)) * 16 + base;
    BsF[i0] = (_Float16)fb0[it];
    BsF[i1] = (_Float16)fb1[it];
  }
}

// Branch-free (per template) epilogue. Row offsets (ra*16+v)*ldc are
// unroll-time-constant multiples of a uniform value -> scalar address math.
template <bool ACC, bool BIASRELU, bool FULL>
__device__ __forceinline__
void sieg_epilogue(float* __restrict__ cb, size_t ldcS, int mBase, int M,
                   const v8f (&acc)[2][2], float bv0, float bv1) {
  float oldv[2][2][8];
  if (ACC) {
#pragma unroll
    for (int ra = 0; ra < 2; ++ra)
#pragma unroll
      for (int v = 0; v < 8; ++v) {
        const float* cp = cb + (size_t)(ra * 16 + v) * ldcS;
        bool ok = FULL || (mBase + ra * 16 + v) < M;
        oldv[ra][0][v] = ok ? cp[0]  : 0.0f;
        oldv[ra][1][v] = ok ? cp[16] : 0.0f;
      }
  }
#pragma unroll
  for (int ra = 0; ra < 2; ++ra)
#pragma unroll
    for (int v = 0; v < 8; ++v) {
      if (!FULL && (mBase + ra * 16 + v) >= M) continue;
      float* cp = cb + (size_t)(ra * 16 + v) * ldcS;
      float val0 = acc[ra][0][v];
      float val1 = acc[ra][1][v];
      if (ACC) { val0 += oldv[ra][0][v]; val1 += oldv[ra][1][v]; }
      if (BIASRELU) {
        val0 = fmaxf(val0 + bv0, 0.0f);
        val1 = fmaxf(val1 + bv1, 0.0f);
      }
      cp[0]  = val0;
      cp[16] = val1;
    }
}

// ---------------------------------------------------------------------------
// WMMA GEMM: C[M,N] (+)= gather(A)[M,K] @ B[K,N]  (+bias, ReLU)
//   Block tile 128x64, 8 waves; each wave computes 32x32 (2x2 fragments,
//   4 v_wmma per 32-K step). Requires N % 64 == 0, K % 32 == 0; M guarded.
// ---------------------------------------------------------------------------
template <bool GATHER, bool ACC, bool BIASRELU>
__global__ __launch_bounds__(256)
void sieg_gemm_f16_wmma(const float* __restrict__ A, const float* __restrict__ B,
                        const float* __restrict__ bias, float* __restrict__ C,
                        int M, int N, int K, int lda, int ldb, int ldc,
                        const int* __restrict__ rowIdx, int idxStride) {
  __shared__ _Float16 AsF[2][A_FRAG_HALVES];   // 16 KB double-buffered
  __shared__ _Float16 BsF[2][B_FRAG_HALVES];   //  8 KB double-buffered

  const int tileM = blockIdx.y * GEMM_TM;
  const int tileN = blockIdx.x * GEMM_TN;
  const int lane  = threadIdx.x & 31;
  const int wv    = threadIdx.x >> 5;     // 8 waves
  const int wm    = wv & 3;               // 4 row blocks of 32
  const int wn    = wv >> 2;              // 2 col blocks of 32
  const int lrow  = lane & 15;
  const int hf    = lane >> 4;
  const bool fullM = (tileM + GEMM_TM) <= M;

  v8f acc[2][2] = {{{}, {}}, {{}, {}}};

  if (fullM) sieg_stage<GATHER, true >(A, B, AsF[0], BsF[0], tileM, tileN, 0, M, lda, ldb, rowIdx, idxStride);
  else       sieg_stage<GATHER, false>(A, B, AsF[0], BsF[0], tileM, tileN, 0, M, lda, ldb, rowIdx, idxStride);
  __syncthreads();

  int cur = 0;
  for (int k0 = 0; k0 < K; k0 += GEMM_TK) {
    const _Float16* Ab = AsF[cur];
    const _Float16* Bb = BsF[cur];
    // Fragment loads: 32B contiguous per lane -> 2x ds_load_b128 each.
    v16h a0 = *reinterpret_cast<const v16h*>(&Ab[(((wm * 2 + 0) << 5) + lane) * 16]);
    v16h a1 = *reinterpret_cast<const v16h*>(&Ab[(((wm * 2 + 1) << 5) + lane) * 16]);
    v16h b0 = *reinterpret_cast<const v16h*>(&Bb[(((wn * 2 + 0) << 5) + lane) * 16]);
    v16h b1 = *reinterpret_cast<const v16h*>(&Bb[(((wn * 2 + 1) << 5) + lane) * 16]);

    // Operand reuse chain: a0b0 -> a0b1 (reuse A) -> a1b1 (reuse B) -> a1b0 (reuse A).
    acc[0][0] = __builtin_amdgcn_wmma_f32_16x16x32_f16(false, a0, false, b0, (short)0, acc[0][0], false, false);
    acc[0][1] = __builtin_amdgcn_wmma_f32_16x16x32_f16(false, a0, false, b1, (short)0, acc[0][1], true,  false);
    acc[1][1] = __builtin_amdgcn_wmma_f32_16x16x32_f16(false, a1, false, b1, (short)0, acc[1][1], false, true);
    acc[1][0] = __builtin_amdgcn_wmma_f32_16x16x32_f16(false, a1, false, b0, (short)0, acc[1][0], true,  false);

    // Stage next tile into the other buffer; global loads overlap WMMA (XDL).
    if (k0 + GEMM_TK < K) {
      if (fullM) sieg_stage<GATHER, true >(A, B, AsF[cur ^ 1], BsF[cur ^ 1], tileM, tileN, k0 + GEMM_TK, M, lda, ldb, rowIdx, idxStride);
      else       sieg_stage<GATHER, false>(A, B, AsF[cur ^ 1], BsF[cur ^ 1], tileM, tileN, k0 + GEMM_TK, M, lda, ldb, rowIdx, idxStride);
    }
    __syncthreads();
    cur ^= 1;
  }

  // Epilogue. C/D layout: VGPR v -> m = v + 8*hf, n = lane%16.
  const int mBase = tileM + wm * 32 + 8 * hf;
  const int n0    = tileN + wn * 32 + lrow;
  const size_t ldcS = (size_t)ldc;
  float* cb = C + (size_t)mBase * ldcS + n0;
  float bv0 = 0.0f, bv1 = 0.0f;
  if (BIASRELU) { bv0 = bias[n0]; bv1 = bias[n0 + 16]; }

  if (fullM) sieg_epilogue<ACC, BIASRELU, true >(cb, ldcS, mBase, M, acc, bv0, bv1);
  else       sieg_epilogue<ACC, BIASRELU, false>(cb, ldcS, mBase, M, acc, bv0, bv1);
}

// ---------------------------------------------------------------------------
// Small utility kernels
// ---------------------------------------------------------------------------
__global__ void sieg_zero_f32(float* __restrict__ p, size_t n) {
  size_t i = (size_t)blockIdx.x * blockDim.x + threadIdx.x;
  size_t stride = (size_t)gridDim.x * blockDim.x;
  for (; i < n; i += stride) p[i] = 0.0f;
}

__global__ void sieg_deg_count(const int* __restrict__ dst, float* __restrict__ deg, int E) {
  int i = blockIdx.x * blockDim.x + threadIdx.x;
  if (i < E) atomicAdd(&deg[dst[i]], 1.0f);
}

__global__ void sieg_dinv(const float* __restrict__ deg, float* __restrict__ dinv, int N) {
  int i = blockIdx.x * blockDim.x + threadIdx.x;
  if (i < N) dinv[i] = rsqrtf(deg[i] + 1.0f);   // +1 for the self loop
}

// One wave per edge (E real edges then N self loops): agg[dst] += xw[src]*norm
__global__ __launch_bounds__(256)
void sieg_scatter_agg(const float* __restrict__ xw, float* __restrict__ agg,
                      const int* __restrict__ src, const int* __restrict__ dst,
                      const float* __restrict__ dinv, int E, int N, int F) {
  int w    = (blockIdx.x * blockDim.x + threadIdx.x) >> 5;
  int lane = threadIdx.x & 31;
  if (w >= E + N) return;
  int s, d; float nw;
  if (w < E) { s = src[w]; d = dst[w]; nw = dinv[s] * dinv[d]; }
  else       { s = d = w - E; float di = dinv[s]; nw = di * di; }
  const float* xs = xw  + (size_t)s * F;
  float*       ad = agg + (size_t)d * F;
  for (int f = lane; f < F; f += 32) atomicAdd(&ad[f], xs[f] * nw);
}

// LayerNorm: Y = LN( [pre-ReLU]( X + bias ) ) * g + be  [post-ReLU]
// One 256-thread block per row; F in {256, 512}.
__global__ __launch_bounds__(256)
void sieg_layernorm(const float* __restrict__ X, const float* __restrict__ bias,
                    const float* __restrict__ g, const float* __restrict__ be,
                    float* __restrict__ Y, int F, int preRelu, int postRelu) {
  __shared__ float red[256];
  const int row = blockIdx.x;
  const int per = F >> 8;                  // 1 or 2 elements per thread
  const float* x = X + (size_t)row * F;
  float vals[2];
  float s = 0.0f;
  for (int i = 0; i < per; ++i) {
    int idx = threadIdx.x + (i << 8);
    float t = x[idx] + (bias ? bias[idx] : 0.0f);
    if (preRelu) t = fmaxf(t, 0.0f);
    vals[i] = t;
    s += t;
  }
  red[threadIdx.x] = s; __syncthreads();
  for (int off = 128; off > 0; off >>= 1) {
    if (threadIdx.x < off) red[threadIdx.x] += red[threadIdx.x + off];
    __syncthreads();
  }
  const float mean = red[0] / (float)F;
  __syncthreads();
  float sq = 0.0f;
  for (int i = 0; i < per; ++i) { float dfl = vals[i] - mean; sq += dfl * dfl; }
  red[threadIdx.x] = sq; __syncthreads();
  for (int off = 128; off > 0; off >>= 1) {
    if (threadIdx.x < off) red[threadIdx.x] += red[threadIdx.x + off];
    __syncthreads();
  }
  const float inv = rsqrtf(red[0] / (float)F + 1e-5f);
  float* y = Y + (size_t)row * F;
  for (int i = 0; i < per; ++i) {
    int idx = threadIdx.x + (i << 8);
    float o = (vals[i] - mean) * inv * g[idx] + be[idx];
    if (postRelu) o = fmaxf(o, 0.0f);
    y[idx] = o;
  }
}

__device__ __forceinline__ float sieg_feat_mlp(float v, const float* __restrict__ w1,
                                               const float* __restrict__ b1,
                                               const float* __restrict__ w2,
                                               const float* __restrict__ b2) {
  float s = 0.0f;
#pragma unroll
  for (int i = 0; i < 16; ++i) s += fmaxf(v * w1[i] + b1[i], 0.0f) * w2[i];
  return s + b2[0];
}

// Per pair: gate = sigmoid(dot(Q,K)/16 + S);  V *= gate  (in place -> attn)
__global__ __launch_bounds__(256)
void sieg_pair_gate(const float* __restrict__ Q, const float* __restrict__ K,
                    float* __restrict__ V,
                    const int* __restrict__ spd, const float* __restrict__ cn,
                    const float* __restrict__ aa, const float* __restrict__ jac,
                    const float* __restrict__ spd_tab,
                    const float* __restrict__ cw1, const float* __restrict__ cb1,
                    const float* __restrict__ cw2, const float* __restrict__ cb2,
                    const float* __restrict__ aw1, const float* __restrict__ ab1,
                    const float* __restrict__ aw2, const float* __restrict__ ab2,
                    const float* __restrict__ jw1, const float* __restrict__ jb1,
                    const float* __restrict__ jw2, const float* __restrict__ jb2,
                    int P, int F) {
  int p    = (blockIdx.x * blockDim.x + threadIdx.x) >> 5;
  int lane = threadIdx.x & 31;
  if (p >= P) return;
  const float* q = Q + (size_t)p * F;
  const float* k = K + (size_t)p * F;
  float dot = 0.0f;
  for (int f = lane; f < F; f += 32) dot += q[f] * k[f];
#pragma unroll
  for (int m = 16; m > 0; m >>= 1) dot += __shfl_xor(dot, m, 32);
  float A = dot * 0.0625f;                       // 1/sqrt(256)
  int sp = spd[p]; sp = sp < 0 ? 0 : (sp > 6 ? 6 : sp);
  float S = spd_tab[sp]
          + sieg_feat_mlp(cn[p],  cw1, cb1, cw2, cb2)
          + sieg_feat_mlp(aa[p],  aw1, ab1, aw2, ab2)
          + sieg_feat_mlp(jac[p], jw1, jb1, jw2, jb2);
  float gate = 1.0f / (1.0f + __expf(-(A + S)));
  float* v = V + (size_t)p * F;
  for (int f = lane; f < F; f += 32) v[f] *= gate;
}

// out[p] = dot(Z[p,:256], w) + b   — one wave per pair
__global__ __launch_bounds__(256)
void sieg_final_dot(const float* __restrict__ Z, const float* __restrict__ w,
                    const float* __restrict__ b, float* __restrict__ out, int P, int F) {
  int p    = (blockIdx.x * blockDim.x + threadIdx.x) >> 5;
  int lane = threadIdx.x & 31;
  if (p >= P) return;
  const float* z = Z + (size_t)p * F;
  float dot = 0.0f;
  for (int f = lane; f < F; f += 32) dot += z[f] * w[f];
#pragma unroll
  for (int m = 16; m > 0; m >>= 1) dot += __shfl_xor(dot, m, 32);
  if (lane == 0) out[p] = dot + b[0];
}

// ---------------------------------------------------------------------------
// Host-side orchestration
// ---------------------------------------------------------------------------
template <bool GATHER, bool ACC, bool BIASRELU>
static inline void launch_gemm(const float* A, const float* B, const float* bias, float* C,
                               int M, int N, int K, const int* rowIdx, int idxStride,
                               hipStream_t stream) {
  dim3 grid(N / GEMM_TN, (M + GEMM_TM - 1) / GEMM_TM);
  sieg_gemm_f16_wmma<GATHER, ACC, BIASRELU><<<grid, 256, 0, stream>>>(
      A, B, bias, C, M, N, K, K /*lda*/, N /*ldb*/, N /*ldc*/, rowIdx, idxStride);
}

extern "C" void kernel_launch(void* const* d_in, const int* in_sizes, int n_in,
                              void* d_out, int out_size, void* d_ws, size_t ws_size,
                              hipStream_t stream) {
  (void)n_in; (void)out_size; (void)ws_size;
  const int IN = 1024, HID = 512, OUT = 256;
  const int N = in_sizes[0] / IN;
  const int E = in_sizes[1] / 2;
  const int P = in_sizes[3];

  const float* x     = (const float*)d_in[0];
  const int*   eidx  = (const int*)  d_in[1];
  const int*   pairs = (const int*)  d_in[2];
  const int*   spd   = (const int*)  d_in[3];
  const float* cn    = (const float*)d_in[4];
  const float* aa    = (const float*)d_in[5];
  const float* jac   = (const float*)d_in[6];
  const float* W1    = (const float*)d_in[7];
  const float* b1    = (const float*)d_in[8];
  const float* g1    = (const float*)d_in[9];
  const float* be1   = (const float*)d_in[10];
  const float* W2    = (const float*)d_in[11];
  const float* b2    = (const float*)d_in[12];
  const float* g2    = (const float*)d_in[13];
  const float* be2   = (const float*)d_in[14];
  const float* W3    = (const float*)d_in[15];
  const float* b3    = (const float*)d_in[16];
  const float* g3    = (const float*)d_in[17];
  const float* be3   = (const float*)d_in[18];
  const float* Wskip = (const float*)d_in[19];
  const float* WQ    = (const float*)d_in[20];
  const float* WK    = (const float*)d_in[21];
  const float* WV    = (const float*)d_in[22];
  const float* spdT  = (const float*)d_in[23];
  const float* cw1   = (const float*)d_in[24];
  const float* cb1   = (const float*)d_in[25];
  const float* cw2   = (const float*)d_in[26];
  const float* cb2   = (const float*)d_in[27];
  const float* aw1   = (const float*)d_in[28];
  const float* ab1   = (const float*)d_in[29];
  const float* aw2   = (const float*)d_in[30];
  const float* ab2   = (const float*)d_in[31];
  const float* jw1   = (const float*)d_in[32];
  const float* jb1   = (const float*)d_in[33];
  const float* jw2   = (const float*)d_in[34];
  const float* jb2   = (const float*)d_in[35];
  const float* m_w1  = (const float*)d_in[36];
  const float* m_b1  = (const float*)d_in[37];
  const float* m_g   = (const float*)d_in[38];
  const float* m_be  = (const float*)d_in[39];
  const float* m_w2  = (const float*)d_in[40];
  const float* m_b2  = (const float*)d_in[41];
  const float* m_w3  = (const float*)d_in[42];
  const float* m_b3  = (const float*)d_in[43];
  float* out = (float*)d_out;

  const int* src = eidx;        // edge_index[0]
  const int* dst = eidx + E;    // edge_index[1]

  // Workspace layout (floats)
  float* ws   = (float*)d_ws;
  float* dinv = ws;                    ws += N;
  float* deg  = ws;                    ws += N;
  float* bufA = ws;                    ws += (size_t)N * HID;   // GEMM outputs (h@W)
  float* bufB = ws;                    ws += (size_t)N * HID;   // scatter accumulators
  float* h    = ws;                    ws += (size_t)N * HID;   // layer outputs (h3: N x 256)
  float* Qb   = ws;                    ws += (size_t)P * OUT;
  float* Kb   = ws;                    ws += (size_t)P * OUT;
  float* Vb   = ws;                    ws += (size_t)P * OUT;
  float* out1 = Qb;                    // P x 512 — overlays dead Q,K after gate kernel
  float* z2   = Vb;                    // P x 256 — overlays V after m_w1 GEMMs

  const int scatterBlocks = (E + N + 7) / 8;   // 8 waves / 256-thread block
  const int waveBlocksP   = (P + 7) / 8;

  // ---- Degree / normalization -------------------------------------------
  sieg_zero_f32<<<(N + 255) / 256, 256, 0, stream>>>(deg, (size_t)N);
  sieg_deg_count<<<(E + 255) / 256, 256, 0, stream>>>(dst, deg, E);
  sieg_dinv<<<(N + 255) / 256, 256, 0, stream>>>(deg, dinv, N);

  // ---- GCN layer 1: h = LN(relu(seg(x@W1) + b1)) ------------------------
  launch_gemm<false, false, false>(x, W1, nullptr, bufA, N, HID, IN, nullptr, 0, stream);
  sieg_zero_f32<<<4096, 256, 0, stream>>>(bufB, (size_t)N * HID);
  sieg_scatter_agg<<<scatterBlocks, 256, 0, stream>>>(bufA, bufB, src, dst, dinv, E, N, HID);
  sieg_layernorm<<<N, 256, 0, stream>>>(bufB, b1, g1, be1, h, HID, 1, 0);

  // ---- GCN layer 2 -------------------------------------------------------
  launch_gemm<false, false, false>(h, W2, nullptr, bufA, N, HID, HID, nullptr, 0, stream);
  sieg_zero_f32<<<4096, 256, 0, stream>>>(bufB, (size_t)N * HID);
  sieg_scatter_agg<<<scatterBlocks, 256, 0, stream>>>(bufA, bufB, src, dst, dinv, E, N, HID);
  sieg_layernorm<<<N, 256, 0, stream>>>(bufB, b2, g2, be2, h, HID, 1, 0);

  // ---- GCN layer 3 + skip: h3 = LN(seg(h@W3) + x@Wskip + b3) ------------
  launch_gemm<false, false, false>(h, W3, nullptr, bufA, N, OUT, HID, nullptr, 0, stream);
  sieg_zero_f32<<<4096, 256, 0, stream>>>(bufB, (size_t)N * OUT);
  sieg_scatter_agg<<<scatterBlocks, 256, 0, stream>>>(bufA, bufB, src, dst, dinv, E, N, OUT);
  launch_gemm<false, true, false>(x, Wskip, nullptr, bufB, N, OUT, IN, nullptr, 0, stream);
  sieg_layernorm<<<N, 256, 0, stream>>>(bufB, b3, g3, be3, h, OUT, 0, 0);   // h = h3 (N x 256)

  // ---- Pair head: Q/K/V via gathered WMMA GEMMs -------------------------
  launch_gemm<true, false, false>(h, WQ, nullptr, Qb, P, OUT, OUT, pairs + 0, 2, stream);
  launch_gemm<true, false, false>(h, WK, nullptr, Kb, P, OUT, OUT, pairs + 1, 2, stream);
  launch_gemm<true, false, false>(h, WV, nullptr, Vb, P, OUT, OUT, pairs + 1, 2, stream);
  sieg_pair_gate<<<waveBlocksP, 256, 0, stream>>>(Qb, Kb, Vb, spd, cn, aa, jac, spdT,
                                                  cw1, cb1, cw2, cb2, aw1, ab1, aw2, ab2,
                                                  jw1, jb1, jw2, jb2, P, OUT);

  // ---- zc @ m_w1 without materializing zc: three 256-row panels ----------
  launch_gemm<true,  false, false>(h,  m_w1,                         nullptr, out1, P, HID, OUT, pairs + 0, 2, stream);
  launch_gemm<true,  true,  false>(h,  m_w1 + (size_t)OUT * HID,     nullptr, out1, P, HID, OUT, pairs + 1, 2, stream);
  launch_gemm<false, true,  false>(Vb, m_w1 + (size_t)2 * OUT * HID, nullptr, out1, P, HID, OUT, nullptr, 0, stream);
  sieg_layernorm<<<P, 256, 0, stream>>>(out1, m_b1, m_g, m_be, out1, HID, 0, 1);

  // ---- z2 = relu(out1 @ m_w2 + m_b2); final dot --------------------------
  launch_gemm<false, false, true>(out1, m_w2, m_b2, z2, P, OUT, HID, nullptr, 0, stream);
  sieg_final_dot<<<waveBlocksP, 256, 0, stream>>>(z2, m_w3, m_b3, out, P, OUT);
}
